// Evolution_87058987090518
// MI455X (gfx1250) — matrix operand order repaired and checked by reading, not compiled, not run on hardware
//
#include <hip/hip_runtime.h>
#include <stdint.h>

// ---------------------------------------------------------------------------
// CDNA5 (gfx1250) wave32 WMMA implementation of the snake evolution network.
//   bf16 activations / weights, f32 accumulation, v_wmma_f32_16x16x32_bf16,
//   async global->LDS staging (ASYNCcnt) for all GEMM B-tiles.
// ---------------------------------------------------------------------------

typedef __attribute__((ext_vector_type(16))) __bf16 v16bf;
typedef __attribute__((ext_vector_type(8)))  float  v8f;

#define NPOLY 256
#define PTS   128
#define CH    128

__device__ __forceinline__ unsigned short f2bf(float f) {
  unsigned u = __builtin_bit_cast(unsigned, f);
  unsigned r = u + 0x7FFFu + ((u >> 16) & 1u);   // round-to-nearest-even
  return (unsigned short)(r >> 16);
}
__device__ __forceinline__ float bf2f(unsigned short b) {
  unsigned u = ((unsigned)b) << 16;
  return __builtin_bit_cast(float, u);
}

// ---- CDNA5 async global->LDS copy (tracked by ASYNCcnt) -------------------
// Per ISA 15.18.3: GLOBAL_LOAD_ASYNC_TO_LDS_B128  vdst(lds addr), vaddr, off
// Flat LDS pointers map to LDS by their low 32 bits (aperture truncation).
__device__ __forceinline__ void async_g2l_b128(void* lds_ptr, const void* gptr) {
  unsigned l = (unsigned)(uintptr_t)lds_ptr;
  unsigned long long g = (unsigned long long)(uintptr_t)gptr;
  asm volatile("global_load_async_to_lds_b128 %0, %1, off"
               :: "v"(l), "v"(g) : "memory");
}
__device__ __forceinline__ void wait_async0() {
  asm volatile("s_wait_asynccnt 0" ::: "memory");
}

// ---------------------------------------------------------------------------
// Repack f32 conv weights (O, Csrc, taps) into WMMA A-fragment order, bf16.
// K-index = tap * Cpad + c (tap-major). 16x32 bf16 A layout used here (B matches):
//   m = lane&15 ; K-group base = (lane>>4)*16 ; e -> K[base + e] (contiguous).
// dst[((ks*Mtiles + mt)*32 + lane)*16 + e]
// ---------------------------------------------------------------------------
__global__ __launch_bounds__(256) void repack_kernel(
    const float* __restrict__ W, unsigned short* __restrict__ dst,
    int Mtiles, int Ksteps, int taps, int Cpad, int Cvalid, int srcC, int srcCoff)
{
  int idx = blockIdx.x * blockDim.x + threadIdx.x;
  int total = Ksteps * Mtiles * 32;
  if (idx >= total) return;
  int lane = idx & 31;
  int mt   = (idx >> 5) % Mtiles;
  int ks   = idx / (Mtiles * 32);
  int m    = mt * 16 + (lane & 15);
  int kb   = (lane >> 4) * 16;
  unsigned short* o = dst + (size_t)idx * 16;
#pragma unroll
  for (int e = 0; e < 16; ++e) {
    int kk = kb + e;
    int k  = ks * 32 + kk;
    int t  = k / Cpad;
    int c  = k % Cpad;
    float v = 0.f;
    if (c < Cvalid && t < taps)
      v = W[((size_t)m * srcC + (c + srcCoff)) * taps + t];
    o[e] = f2bf(v);
  }
}

// ---------------------------------------------------------------------------
// Bilinear gather + canonical coords -> bf16 input [N][P][128] (66 real ch).
// grid_sample identity: ix = px - 0.5, iy = py - 0.5.
// ---------------------------------------------------------------------------
__global__ __launch_bounds__(128) void gather_kernel(
    const float* __restrict__ feat,   // (8,64,128,128)
    const float* __restrict__ polys,  // (256,128,2)
    const int*   __restrict__ ind,    // (256,)
    const float* __restrict__ curPy,
    float* __restrict__ pyWork,
    unsigned short* __restrict__ inp,
    int iter)
{
  int n = blockIdx.x, p = threadIdx.x;
  float px, py;
  if (iter == 0) {
    px = polys[(n * PTS + p) * 2 + 0] * 0.25f;
    py = polys[(n * PTS + p) * 2 + 1] * 0.25f;
    px = fminf(fmaxf(px, 0.f), 127.f);
    py = fminf(fmaxf(py, 0.f), 127.f);
  } else {
    px = curPy[(n * PTS + p) * 2 + 0] * 0.25f;
    py = curPy[(n * PTS + p) * 2 + 1] * 0.25f;
  }
  pyWork[(n * PTS + p) * 2 + 0] = px;
  pyWork[(n * PTS + p) * 2 + 1] = py;

  __shared__ float sx[PTS], sy[PTS];
  sx[p] = px; sy[p] = py;
  __syncthreads();
  for (int s = 64; s > 0; s >>= 1) {
    if (p < s) { sx[p] = fminf(sx[p], sx[p + s]); sy[p] = fminf(sy[p], sy[p + s]); }
    __syncthreads();
  }
  float mnx = sx[0], mny = sy[0];

  int b = ind[n];
  float ix = px - 0.5f, iy = py - 0.5f;
  float x0f = floorf(ix), y0f = floorf(iy);
  int   x0 = (int)x0f, y0 = (int)y0f;
  float wx1 = ix - x0f, wy1 = iy - y0f;
  float wq[4] = { (1.f - wx1) * (1.f - wy1), wx1 * (1.f - wy1),
                  (1.f - wx1) * wy1,         wx1 * wy1 };
  int xi[4] = { x0, x0 + 1, x0, x0 + 1 };
  int yi[4] = { y0, y0, y0 + 1, y0 + 1 };
  size_t base[4];
  float  wv[4];
#pragma unroll
  for (int q = 0; q < 4; ++q) {
    bool valid = (xi[q] >= 0) && (xi[q] < 128) && (yi[q] >= 0) && (yi[q] < 128);
    int xc = min(max(xi[q], 0), 127), yc = min(max(yi[q], 0), 127);
    base[q] = (((size_t)b * 64) * 128 + yc) * 128 + xc;
    wv[q] = valid ? wq[q] : 0.f;
  }
  unsigned short* op = inp + ((size_t)n * PTS + p) * CH;
  for (int c = 0; c < 64; ++c) {
    size_t cs = (size_t)c * 128 * 128;
    float acc = feat[base[0] + cs] * wv[0] + feat[base[1] + cs] * wv[1] +
                feat[base[2] + cs] * wv[2] + feat[base[3] + cs] * wv[3];
    op[c] = f2bf(acc);
  }
  op[64] = f2bf((px - mnx) * 4.0f);
  op[65] = f2bf((py - mny) * 4.0f);
#pragma unroll
  for (int c = 66; c < CH; ++c) op[c] = 0;
}

// ---------------------------------------------------------------------------
// Circular dilated conv (k=9, Cin padded to 128, Cout=128) as WMMA GEMM.
// One workgroup per instance; async-staged LDS tile [128 pts][128 ch] bf16.
// Wave w -> out channels 16w..16w+15; 8 N-tiles of 16 points each.
// Epilogue: +bias, relu, BN affine, optional residual; bf16 store to state.
// ---------------------------------------------------------------------------
__global__ __launch_bounds__(256) void conv9_wmma_kernel(
    const unsigned short* __restrict__ src, int srcStride, int srcCoff,
    unsigned short* __restrict__ dst, int dstStride, int dstCoff,
    const unsigned short* __restrict__ Apack,   // [36][8][32][16]
    const float* __restrict__ bias, const float* __restrict__ gw,
    const float* __restrict__ bbeta, const float* __restrict__ bmean,
    const float* __restrict__ bvar,
    int dilation, int residual)
{
  __shared__ unsigned short xs[PTS * CH];   // 32 KB
  int n = blockIdx.x;
  int tid = threadIdx.x;
  {
    int p  = tid >> 1;
    int cb = (tid & 1) * 64;
    const unsigned short* s = src + ((size_t)n * PTS + p) * srcStride + srcCoff + cb;
    unsigned short* l = xs + p * CH + cb;
#pragma unroll
    for (int j = 0; j < 8; ++j)               // 8 x b128 async per thread
      async_g2l_b128(l + j * 8, s + j * 8);
  }
  wait_async0();
  __syncthreads();

  int wave = tid >> 5, lane = tid & 31;
  int half = lane >> 4, lcol = lane & 15;
  int m0 = wave * 16;

  v8f acc[8];
#pragma unroll
  for (int i = 0; i < 8; ++i)
#pragma unroll
    for (int j = 0; j < 8; ++j) acc[i][j] = 0.f;

  for (int ksI = 0; ksI < 36; ++ksI) {
    v16bf a = *reinterpret_cast<const v16bf*>(
        Apack + (((size_t)ksI * 8 + wave) * 32 + lane) * 16);
    if (ksI + 1 < 36)
      __builtin_prefetch(Apack + (((size_t)(ksI + 1) * 8 + wave) * 32 + lane) * 16, 0, 0);
    int t  = ksI >> 2;             // tap (K is tap-major, 128 ch per tap)
    int c0 = (ksI & 3) * 32;       // channel offset within tap
    int shift = (t - 4) * dilation;
#pragma unroll
    for (int nt = 0; nt < 8; ++nt) {
      int p = ((nt * 16 + lcol) + shift + 128) & 127;   // circular
      v16bf bfr = *reinterpret_cast<const v16bf*>(xs + p * CH + c0 + half * 16);
      acc[nt] = __builtin_amdgcn_wmma_f32_16x16x32_bf16(
          false, a, false, bfr, (short)0, acc[nt], false, false);
    }
  }

  float bs[8], sc[8], mn[8], bt[8];
#pragma unroll
  for (int e = 0; e < 8; ++e) {
    int m = m0 + half * 8 + e;
    bs[e] = bias[m];
    sc[e] = gw[m] * rsqrtf(bvar[m] + 1e-5f);
    mn[e] = bmean[m];
    bt[e] = bbeta[m];
  }
#pragma unroll
  for (int nt = 0; nt < 8; ++nt) {
    int p = nt * 16 + lcol;
    unsigned short ov[8];
#pragma unroll
    for (int e = 0; e < 8; ++e) {
      float v = acc[nt][e] + bs[e];
      v = fmaxf(v, 0.f);
      v = (v - mn[e]) * sc[e] + bt[e];
      if (residual) v += bf2f(xs[p * CH + m0 + half * 8 + e]);
      ov[e] = f2bf(v);
    }
    unsigned w0 = (unsigned)ov[0] | ((unsigned)ov[1] << 16);
    unsigned w1 = (unsigned)ov[2] | ((unsigned)ov[3] << 16);
    unsigned w2 = (unsigned)ov[4] | ((unsigned)ov[5] << 16);
    unsigned w3 = (unsigned)ov[6] | ((unsigned)ov[7] << 16);
    *reinterpret_cast<uint4*>(
        dst + ((size_t)n * PTS + p) * dstStride + dstCoff + m0 + half * 8) =
        make_uint4(w0, w1, w2, w3);
  }
}

// ---------------------------------------------------------------------------
// GEMM epilogue for one 16xP tile: max-reduce (fusion) or bf16 store.
// ---------------------------------------------------------------------------
__device__ __forceinline__ void gemm_epilogue(
    const v8f* acc, int n, int m0, int half, int lcol,
    const float* __restrict__ bias,
    unsigned short* __restrict__ dstBf, int dstStride,
    float* __restrict__ gmaxOut, int relu, int M)
{
  if (gmaxOut) {
#pragma unroll
    for (int e = 0; e < 8; ++e) {
      int m = m0 + half * 8 + e;
      float mx = acc[0][e];
#pragma unroll
      for (int nt = 1; nt < 8; ++nt) mx = fmaxf(mx, acc[nt][e]);
      for (int s = 1; s < 16; s <<= 1)
        mx = fmaxf(mx, __shfl_xor(mx, s, 32));   // stays within 16-lane group
      if (lcol == 0) gmaxOut[(size_t)n * M + m] = mx + bias[m];
    }
  } else {
    float bs[8];
#pragma unroll
    for (int e = 0; e < 8; ++e) bs[e] = bias[m0 + half * 8 + e];
#pragma unroll
    for (int nt = 0; nt < 8; ++nt) {
      int p = nt * 16 + lcol;
      unsigned short ov[8];
#pragma unroll
      for (int e = 0; e < 8; ++e) {
        float v = acc[nt][e] + bs[e];
        if (relu) v = fmaxf(v, 0.f);
        ov[e] = f2bf(v);
      }
      unsigned w0 = (unsigned)ov[0] | ((unsigned)ov[1] << 16);
      unsigned w1 = (unsigned)ov[2] | ((unsigned)ov[3] << 16);
      unsigned w2 = (unsigned)ov[4] | ((unsigned)ov[5] << 16);
      unsigned w3 = (unsigned)ov[6] | ((unsigned)ov[7] << 16);
      *reinterpret_cast<uint4*>(
          dstBf + ((size_t)n * PTS + p) * dstStride + m0 + half * 8) =
          make_uint4(w0, w1, w2, w3);
    }
  }
}

// ---------------------------------------------------------------------------
// 1x1 conv (GEMM) with WMMA. B staged into LDS in 128-channel chunks via
// async loads; each wave owns up to TWO 16-row M-tiles (mt, mt+8) so every
// LDS B-fragment feeds two WMMAs. Global B traffic = exactly 1x.
// Modes: gmaxOut -> max over points (fusion); dstBf -> (+bias, relu?) bf16;
//        vecinit -> per-(n,m) initial accumulator (pred1 global half).
// ---------------------------------------------------------------------------
__global__ __launch_bounds__(256) void gemm1x1_wmma_kernel(
    const unsigned short* __restrict__ Bsrc, int Bstride,
    const unsigned short* __restrict__ Apack, int Mtiles, int Ksteps,
    const float* __restrict__ bias,
    const float* __restrict__ vecinit,
    unsigned short* __restrict__ dstBf, int dstStride,
    float* __restrict__ gmaxOut, int relu, int M)
{
  __shared__ unsigned short xs[PTS * CH];   // 32 KB chunk: [128 pts][128 ch]
  int n = blockIdx.x, tid = threadIdx.x;
  int wave = tid >> 5, lane = tid & 31;
  int half = lane >> 4, lcol = lane & 15;
  const unsigned short* brow = Bsrc + (size_t)n * PTS * Bstride;

  int mt0 = wave, mt1 = wave + 8;
  bool t0v = mt0 < Mtiles, t1v = mt1 < Mtiles;
  int mt0c = t0v ? mt0 : 0;

  v8f acc0[8], acc1[8];
#pragma unroll
  for (int e = 0; e < 8; ++e) {
    float v0 = (vecinit && t0v) ? vecinit[(size_t)n * M + mt0 * 16 + half * 8 + e] : 0.f;
    float v1 = (vecinit && t1v) ? vecinit[(size_t)n * M + mt1 * 16 + half * 8 + e] : 0.f;
#pragma unroll
    for (int nt = 0; nt < 8; ++nt) { acc0[nt][e] = v0; acc1[nt][e] = v1; }
  }

  int sp  = tid >> 1;
  int scb = (tid & 1) * 64;
  unsigned short* lstage = xs + sp * CH + scb;
  const unsigned short* gstage = brow + (size_t)sp * Bstride + scb;

  int Kchunks = Ksteps >> 2;                 // 128 channels per chunk
  for (int kc = 0; kc < Kchunks; ++kc) {
#pragma unroll
    for (int j = 0; j < 8; ++j)              // 8 x b128 async per thread
      async_g2l_b128(lstage + j * 8, gstage + kc * CH + j * 8);
    wait_async0();
    __syncthreads();

    for (int ks = 0; ks < 4; ++ks) {
      int kg = kc * 4 + ks;
      v16bf a0 = *reinterpret_cast<const v16bf*>(
          Apack + (((size_t)kg * Mtiles + mt0c) * 32 + lane) * 16);
      v16bf a1 = t1v ? *reinterpret_cast<const v16bf*>(
          Apack + (((size_t)kg * Mtiles + mt1) * 32 + lane) * 16) : a0;
#pragma unroll
      for (int nt = 0; nt < 8; ++nt) {
        int p = nt * 16 + lcol;
        v16bf bfr = *reinterpret_cast<const v16bf*>(xs + p * CH + ks * 32 + half * 16);
        if (t0v)
          acc0[nt] = __builtin_amdgcn_wmma_f32_16x16x32_bf16(
              false, a0, false, bfr, (short)0, acc0[nt], false, false);
        if (t1v)
          acc1[nt] = __builtin_amdgcn_wmma_f32_16x16x32_bf16(
              false, a1, false, bfr, (short)0, acc1[nt], false, false);
      }
    }
    __syncthreads();                         // protect LDS before next stage
  }

  if (t0v) gemm_epilogue(acc0, n, mt0 * 16, half, lcol, bias, dstBf, dstStride,
                         gmaxOut, relu, M);
  if (t1v) gemm_epilogue(acc1, n, mt1 * 16, half, lcol, bias, dstBf, dstStride,
                         gmaxOut, relu, M);
}

// pred1 "global" half: vec[n][o] = sum_{c<256} w1[o][c] * gmax[n][c]
__global__ __launch_bounds__(256) void pred1vec_kernel(
    const float* __restrict__ w1,    // (256,1280)
    const float* __restrict__ gmax,  // [N][256]
    float* __restrict__ vec)
{
  __shared__ float g[256];
  int n = blockIdx.x, o = threadIdx.x;
  g[o] = gmax[(size_t)n * 256 + o];
  __syncthreads();
  float acc = 0.f;
  const float* wr = w1 + (size_t)o * 1280;
  for (int c = 0; c < 256; ++c) acc += wr[c] * g[c];
  vec[(size_t)n * 256 + o] = acc;
}

// final 64->2 conv + coordinate update + output write
__global__ __launch_bounds__(128) void pred3_update_kernel(
    const unsigned short* __restrict__ act2,  // [N][P][64] bf16
    const float* __restrict__ w3, const float* __restrict__ b3,
    const float* __restrict__ pyWork,
    float* __restrict__ curPy, float* __restrict__ outSlab)
{
  int n = blockIdx.x, p = threadIdx.x;
  const unsigned short* a = act2 + ((size_t)n * PTS + p) * 64;
  float o0 = b3[0], o1 = b3[1];
  for (int c = 0; c < 64; ++c) {
    float v = bf2f(a[c]);
    o0 += w3[c] * v;
    o1 += w3[64 + c] * v;
  }
  float nx = pyWork[(n * PTS + p) * 2 + 0] * 4.f + o0;
  float ny = pyWork[(n * PTS + p) * 2 + 1] * 4.f + o1;
  curPy[(n * PTS + p) * 2 + 0] = nx;
  curPy[(n * PTS + p) * 2 + 1] = ny;
  outSlab[(n * PTS + p) * 2 + 0] = nx;
  outSlab[(n * PTS + p) * 2 + 1] = ny;
}

// ---------------------------------------------------------------------------
extern "C" void kernel_launch(void* const* d_in, const int* in_sizes, int n_in,
                              void* d_out, int out_size, void* d_ws, size_t ws_size,
                              hipStream_t stream) {
  (void)in_sizes; (void)n_in; (void)out_size; (void)ws_size;

  char* ws = (char*)d_ws;
  size_t off = 0;
  auto take = [&](size_t bytes) -> char* {
    char* p = ws + off;
    off += (bytes + 255) & ~(size_t)255;
    return p;
  };

  unsigned short* inp   = (unsigned short*)take((size_t)NPOLY * PTS * CH * 2);
  unsigned short* state = (unsigned short*)take((size_t)NPOLY * PTS * 1024 * 2);
  unsigned short* act1  = (unsigned short*)take((size_t)NPOLY * PTS * 256 * 2);
  unsigned short* act2  = (unsigned short*)take((size_t)NPOLY * PTS * 64 * 2);
  float* pyWork = (float*)take((size_t)NPOLY * PTS * 2 * 4);
  float* curPy  = (float*)take((size_t)NPOLY * PTS * 2 * 4);
  float* gmax   = (float*)take((size_t)NPOLY * 256 * 4);
  float* vec    = (float*)take((size_t)NPOLY * 256 * 4);

  unsigned short* packHead[3];
  unsigned short* packRes[3][7];
  unsigned short* packFus[3];
  unsigned short* packP1[3];
  unsigned short* packP2[3];
  for (int s = 0; s < 3; ++s) {
    packHead[s] = (unsigned short*)take((size_t)36 * 8 * 32 * 16 * 2);
    for (int i = 0; i < 7; ++i)
      packRes[s][i] = (unsigned short*)take((size_t)36 * 8 * 32 * 16 * 2);
    packFus[s] = (unsigned short*)take((size_t)32 * 16 * 32 * 16 * 2);
    packP1[s]  = (unsigned short*)take((size_t)32 * 16 * 32 * 16 * 2);
    packP2[s]  = (unsigned short*)take((size_t)8 * 4 * 32 * 16 * 2);
  }

  const float* feat  = (const float*)d_in[0];
  const float* polys = (const float*)d_in[1];
  const int*   ind   = (const int*)d_in[2];
  auto P = [&](int s, int j) -> const float* {
    return (const float*)d_in[3 + s * 56 + j];
  };
  // per-snake leaf order (insertion order of init_snake):
  //  head: w=0 b=1 g=2 beta=3 m=4 v=5
  //  res i: w=6+6i b=7+6i g=8+6i beta=9+6i m=10+6i v=11+6i
  //  fusion: w=48 b=49 ; pred: w1=50 b1=51 w2=52 b2=53 w3=54 b3=55

  // --- repack all weights into WMMA A-fragment order (bf16) ---
  for (int s = 0; s < 3; ++s) {
    int tot = 36 * 8 * 32;
    repack_kernel<<<(tot + 255) / 256, 256, 0, stream>>>(
        P(s, 0), packHead[s], 8, 36, 9, 128, 66, 66, 0);
    for (int i = 0; i < 7; ++i)
      repack_kernel<<<(tot + 255) / 256, 256, 0, stream>>>(
          P(s, 6 + 6 * i), packRes[s][i], 8, 36, 9, 128, 128, 128, 0);
    tot = 32 * 16 * 32;
    repack_kernel<<<(tot + 255) / 256, 256, 0, stream>>>(
        P(s, 48), packFus[s], 16, 32, 1, 1024, 1024, 1024, 0);
    repack_kernel<<<(tot + 255) / 256, 256, 0, stream>>>(
        P(s, 50), packP1[s], 16, 32, 1, 1024, 1024, 1280, 256);
    tot = 8 * 4 * 32;
    repack_kernel<<<(tot + 255) / 256, 256, 0, stream>>>(
        P(s, 52), packP2[s], 4, 8, 1, 256, 256, 256, 0);
  }

  static const int DILS[7] = {1, 1, 1, 2, 2, 4, 4};

  for (int k = 0; k < 3; ++k) {
    gather_kernel<<<NPOLY, 128, 0, stream>>>(feat, polys, ind, curPy, pyWork, inp, k);

    // head: inp(128-ch padded) -> state slice 0
    conv9_wmma_kernel<<<NPOLY, 256, 0, stream>>>(
        inp, CH, 0, state, 1024, 0, packHead[k],
        P(k, 1), P(k, 2), P(k, 3), P(k, 4), P(k, 5), 1, 0);

    // res blocks: state slice i -> slice i+1 (with residual)
    for (int i = 0; i < 7; ++i)
      conv9_wmma_kernel<<<NPOLY, 256, 0, stream>>>(
          state, 1024, 128 * i, state, 1024, 128 * (i + 1), packRes[k][i],
          P(k, 7 + 6 * i), P(k, 8 + 6 * i), P(k, 9 + 6 * i),
          P(k, 10 + 6 * i), P(k, 11 + 6 * i), DILS[i], 1);

    // fusion 1024->256 + max over points -> gmax
    gemm1x1_wmma_kernel<<<NPOLY, 256, 0, stream>>>(
        state, 1024, packFus[k], 16, 32, P(k, 49),
        nullptr, nullptr, 0, gmax, 0, 256);

    // pred1 global half (W1[:, :256] @ gmax)
    pred1vec_kernel<<<NPOLY, 256, 0, stream>>>(P(k, 50), gmax, vec);

    // pred1 state half: 1024->256 with init=vec, +b1, relu -> act1
    gemm1x1_wmma_kernel<<<NPOLY, 256, 0, stream>>>(
        state, 1024, packP1[k], 16, 32, P(k, 51),
        vec, act1, 256, nullptr, 1, 256);

    // pred2: 256->64, relu -> act2
    gemm1x1_wmma_kernel<<<NPOLY, 256, 0, stream>>>(
        act1, 256, packP2[k], 4, 8, P(k, 53),
        nullptr, act2, 64, nullptr, 1, 64);

    // pred3: 64->2, py update, write output slab k
    pred3_update_kernel<<<NPOLY, 128, 0, stream>>>(
        act2, P(k, 54), P(k, 55), pyWork, curPy,
        (float*)d_out + (size_t)k * NPOLY * PTS * 2);
  }
}
